// ScoreModelHead_50062138802498
// MI455X (gfx1250) — matrix operand order repaired and checked by reading, not compiled, not run on hardware
//
#include <hip/hip_runtime.h>
#include <math.h>

// ---------------- problem constants ----------------
#define NT_   64
#define NQ_   256
#define NN_   (NT_*NQ_)      // 16384
#define NE_   (NN_*16)       // 262144
#define DS_   64
#define DV_   32
#define TE_   128
#define RB_   8
#define FDIM_ 160            // DS + 3*DV
#define HID_  64
#define HOUT_ 192            // DS + 4*DV
#define KPAD_ 160            // 136 -> padded to 5 * 32
#define TILE_E_ 128
#define NTILES_ (NE_/TILE_E_)   // 2048
#define EDGE_GRID_ 512          // persistent blocks; 4 tiles each

typedef __attribute__((ext_vector_type(16))) __bf16 v16bf;
typedef __attribute__((ext_vector_type(8)))  float  v8f;

static __device__ __forceinline__ unsigned short f2bf(float f) {
  union { float f; unsigned u; } v; v.f = f;
  unsigned u = v.u;
  return (unsigned short)((u + 0x7FFFu + ((u >> 16) & 1u)) >> 16);
}

// fast transcendentals on the native TRANS pipe (v_exp_f32 / v_rcp_f32)
static __device__ __forceinline__ float fast_exp(float x) {
  return __builtin_amdgcn_exp2f(x * 1.44269504088896340736f);
}
static __device__ __forceinline__ float fast_sigmoid(float x) {
  return __builtin_amdgcn_rcpf(1.f + fast_exp(-x));
}

// ---------------- workspace layout (all sizes multiples of 256B) ----------------
// total ~21.8 MB
static const size_t OFF_R     = 0;                       // 64*9*4      = 2304
static const size_t OFF_TR    = 2304;                    // 64*3*4      = 768
static const size_t OFF_QEMB  = 3072;                    // 64*128*2    = 16384 (bf16)
static const size_t OFF_QXF   = 19456;                   // 16384*3*4   = 196608
static const size_t OFF_NUM0  = 216064;                  // 16384*160*4 = 10485760
static const size_t OFF_NUM1  = 10701824;                // 10485760
static const size_t OFF_CNT0  = 21187584;                // 16384*4     = 65536
static const size_t OFF_CNT1  = 21253120;                // 65536
static const size_t OFF_W1T   = 21318656;                // 2*64*160*2  = 40960 (bf16, transposed+padded)
static const size_t OFF_W2T   = 21359616;                // 2*192*64*2  = 49152 (bf16, transposed)
static const size_t OFF_LINQ  = 21408768;                // 16384*3*4   = 196608
static const size_t OFF_ANGQ  = 21605376;                // 196608
static const int    ZERO_FLOATS = (int)((OFF_W1T - OFF_NUM0) / 4);  // num0,num1,cnt0,cnt1

// ---------------- prep kernels ----------------
__global__ void k_prep_R(const float* __restrict__ Ts, float* __restrict__ R, float* __restrict__ tr) {
  int t = threadIdx.x;
  if (t >= NT_) return;
  float w = Ts[t*7+0], x = Ts[t*7+1], y = Ts[t*7+2], z = Ts[t*7+3];
  float inv = 1.f / sqrtf(w*w + x*x + y*y + z*z);
  w *= inv; x *= inv; y *= inv; z *= inv;
  float* Rt = R + t*9;
  Rt[0] = 1.f - 2.f*(y*y + z*z); Rt[1] = 2.f*(x*y - w*z);       Rt[2] = 2.f*(x*z + w*y);
  Rt[3] = 2.f*(x*y + w*z);       Rt[4] = 1.f - 2.f*(x*x + z*z); Rt[5] = 2.f*(y*z - w*x);
  Rt[6] = 2.f*(x*z - w*y);       Rt[7] = 2.f*(y*z + w*x);       Rt[8] = 1.f - 2.f*(x*x + y*y);
  tr[t*3+0] = Ts[t*7+4]; tr[t*3+1] = Ts[t*7+5]; tr[t*3+2] = Ts[t*7+6];
}

__global__ void k_prep_qemb(const float* __restrict__ timev, const float* __restrict__ W_qt,
                            const float* __restrict__ b_qt, unsigned short* __restrict__ qembbf) {
  __shared__ float te[TE_];
  int t = blockIdx.x, e = threadIdx.x;
  if (e < TE_/2) {
    float fr = expf(-logf(10000.f) * (float)e / (float)(TE_/2 - 1));
    float a = timev[t] * fr;
    te[e] = sinf(a);
    te[e + TE_/2] = cosf(a);
  }
  __syncthreads();
  float acc = b_qt[e];
  for (int k = 0; k < TE_; ++k) acc += te[k] * W_qt[k*TE_ + e];
  qembbf[t*TE_ + e] = f2bf(acc);
}

__global__ void k_prep_qx(const float* __restrict__ R, const float* __restrict__ tr,
                          const float* __restrict__ qx, float* __restrict__ qxf) {
  int n = blockIdx.x * blockDim.x + threadIdx.x;
  if (n >= NN_) return;
  int t = n >> 8, q = n & 255;
  const float* Rt = R + t*9;
  float x0 = qx[q*3+0], x1 = qx[q*3+1], x2 = qx[q*3+2];
  qxf[n*3+0] = Rt[0]*x0 + Rt[1]*x1 + Rt[2]*x2 + tr[t*3+0];
  qxf[n*3+1] = Rt[3]*x0 + Rt[4]*x1 + Rt[5]*x2 + tr[t*3+1];
  qxf[n*3+2] = Rt[6]*x0 + Rt[7]*x1 + Rt[8]*x2 + tr[t*3+2];
}

// W1t[s][n][k] = bf16(W1_r[s][k][n]), K padded 136 -> 160 with zeros
__global__ void k_prep_w1t(const float* __restrict__ W1_r, unsigned short* __restrict__ w1t) {
  int idx = blockIdx.x * blockDim.x + threadIdx.x;
  if (idx >= 2*HID_*KPAD_) return;
  int s   = idx / (HID_*KPAD_);
  int rem = idx - s*(HID_*KPAD_);
  int n   = rem / KPAD_;
  int k   = rem - n*KPAD_;
  float v = (k < RB_ + TE_) ? W1_r[(s*(RB_+TE_) + k)*HID_ + n] : 0.f;
  w1t[idx] = f2bf(v);
}

// W2t[s][n][k] = bf16(W2_r[s][k][n])
__global__ void k_prep_w2t(const float* __restrict__ W2_r, unsigned short* __restrict__ w2t) {
  int idx = blockIdx.x * blockDim.x + threadIdx.x;
  if (idx >= 2*HOUT_*HID_) return;
  int s   = idx / (HOUT_*HID_);
  int rem = idx - s*(HOUT_*HID_);
  int n   = rem / HID_;
  int k   = rem - n*HID_;
  w2t[idx] = f2bf(W2_r[(s*HID_ + k)*HOUT_ + n]);
}

__global__ void k_zero(float* __restrict__ p, int count) {
  int stride = gridDim.x * blockDim.x;
  for (int i = blockIdx.x * blockDim.x + threadIdx.x; i < count; i += stride) p[i] = 0.f;
}

// ---------------- edge kernel: persistent WMMA MLP + message + scatter ----------------
struct EdgeInfo { float dx, dy, dz; int src; int dst; int pad; };   // 24 B

__global__ __launch_bounds__(256) void k_edge(
    const int* __restrict__ esrc, const int* __restrict__ edst,
    const float* __restrict__ key_x, const float* __restrict__ key_f,
    const float* __restrict__ qxf, const unsigned short* __restrict__ qembbf,
    const unsigned short* __restrict__ w1t, const float* __restrict__ b1g,
    const unsigned short* __restrict__ w2t, const float* __restrict__ b2g,
    float* __restrict__ num, float* __restrict__ cnt)
{
  // 140 KB LDS, phase-aliased: [Xs(40960)|W1s(20480)] overlaid by Y2s(98304) in phase 2
  __shared__ __align__(16) unsigned char smem[143360];
  unsigned short (*Xs)[KPAD_]  = (unsigned short(*)[KPAD_])(smem);
  unsigned short (*W1s)[KPAD_] = (unsigned short(*)[KPAD_])(smem + 40960);
  float          (*Y2s)[HOUT_] = (float(*)[HOUT_])(smem);
  unsigned short (*Hs)[HID_]   = (unsigned short(*)[HID_])(smem + 98304);
  unsigned short (*W2s)[HID_]  = (unsigned short(*)[HID_])(smem + 114688);
  float* b1s = (float*)(smem + 139264);
  float* b2s = (float*)(smem + 139520);
  EdgeInfo* Ed = (EdgeInfo*)(smem + 140288);

  const int tid = threadIdx.x;

  // stage weights ONCE per persistent block (W1s survives only until first Y2s
  // write would clobber it -- but W1s region is re-read every tile, so keep a
  // private reload of just W1s each tile? No: W1s lives at +40960 which IS
  // clobbered by Y2s rows 53..79. Therefore stage W1s into the *non-aliased*
  // W2s/Hs zone? Not enough room. Instead: reload W1s each tile (20 KB from
  // L2), but W2s/b1s/b2s (25 KB) are staged once -- Y2s never touches them.
  {
    const unsigned int* s2 = (const unsigned int*)w2t;
    unsigned int* d2 = (unsigned int*)&W2s[0][0];
    for (int i = tid; i < (HOUT_*HID_)/2; i += 256) d2[i] = s2[i];
    if (tid < HID_)  b1s[tid] = b1g[tid];
    if (tid < HOUT_) b2s[tid] = b2g[tid];
  }

  const int w      = tid >> 5;
  const int lane   = tid & 31;
  const int rowA   = w*16 + (lane & 15);
  const int khalfA = (lane >> 4) * 8;      // A frag: lanes<16 -> K 0..7/16..23; lanes>=16 -> 8..15/24..31
  const int colB   = lane & 15;
  const int kbB    = (lane >> 4) * 8;      // B frag: lanes<16 -> K 0..7/16..23; lanes>=16 -> 8..15/24..31
  const int rbase  = w*16 + ((lane >> 4) << 3);  // C frag rows

  for (int tile = blockIdx.x; tile < NTILES_; tile += gridDim.x) {
    __syncthreads();   // previous tile's message phase done reading Y2s/Ed

    // reload W1s (aliased by Y2s last tile)
    {
      const unsigned int* s1 = (const unsigned int*)w1t;
      unsigned int* d1 = (unsigned int*)&W1s[0][0];
      for (int i = tid; i < (HID_*KPAD_)/2; i += 256) d1[i] = s1[i];
    }

    // per-edge geometry + radial basis into Xs[:,0:8], zero pad Xs[:,136:160]
    if (tid < TILE_E_) {
      int e = tile * TILE_E_ + tid;
      int s = esrc[e], d = edst[e];
      float kx0 = key_x[(size_t)s*3+0], kx1 = key_x[(size_t)s*3+1], kx2 = key_x[(size_t)s*3+2];
      float r0 = kx0 - qxf[(size_t)d*3+0];
      float r1 = kx1 - qxf[(size_t)d*3+1];
      float r2 = kx2 - qxf[(size_t)d*3+2];
      float r  = sqrtf(r0*r0 + r1*r1 + r2*r2);
      float iv = __builtin_amdgcn_rcpf(r + 1e-8f);
      EdgeInfo ei; ei.dx = r0*iv; ei.dy = r1*iv; ei.dz = r2*iv; ei.src = s; ei.dst = d; ei.pad = 0;
      Ed[tid] = ei;
      // prefetch this edge's key_f row (640 B) into cache ahead of the message phase
      const float* kfrow = key_f + (size_t)s * FDIM_;
      __builtin_prefetch(kfrow, 0, 3);
      __builtin_prefetch(kfrow + 128, 0, 3);
      __builtin_prefetch(kfrow + 144, 0, 3);
      #pragma unroll
      for (int j = 0; j < RB_; ++j) {
        float c = 4.f * (float)j / 7.f;
        float dl = r - c;
        Xs[tid][j] = f2bf(fast_exp(-dl*dl*4.f));      // exp(-(r-c)^2/0.25)
      }
      unsigned int* padp = (unsigned int*)&Xs[tid][136];
      #pragma unroll
      for (int j = 0; j < 12; ++j) padp[j] = 0u;
    }
    __syncthreads();

    // copy qemb[t] (bf16) into Xs[:,8:136]
    for (int idx = tid; idx < TILE_E_ * (TE_/2); idx += 256) {
      int e = idx >> 6, j = idx & 63;
      int t = Ed[e].dst >> 8;
      ((unsigned int*)&Xs[e][RB_])[j] = ((const unsigned int*)(qembbf + (size_t)t*TE_))[j];
    }
    __syncthreads();

    // ---- GEMM1: Y1(128x64) = Xs(128x160) @ W1(160x64), silu -> Hs (bf16) ----
    v8f acc1[4];
    #pragma unroll
    for (int nt = 0; nt < 4; ++nt)
      #pragma unroll
      for (int rr = 0; rr < 8; ++rr) acc1[nt][rr] = 0.f;

    for (int kc = 0; kc < 5; ++kc) {
      union { uint4 u[2]; v16bf v; } A;
      A.u[0] = *(const uint4*)&Xs[rowA][kc*32 + khalfA];
      A.u[1] = *(const uint4*)&Xs[rowA][kc*32 + khalfA + 16];
      #pragma unroll
      for (int nt = 0; nt < 4; ++nt) {
        union { uint4 u[2]; v16bf v; } B;
        B.u[0] = *(const uint4*)&W1s[nt*16 + colB][kc*32 + kbB];
        B.u[1] = *(const uint4*)&W1s[nt*16 + colB][kc*32 + kbB + 16];
        acc1[nt] = __builtin_amdgcn_wmma_f32_16x16x32_bf16(
            false, A.v, false, B.v, (short)0, acc1[nt], false, false);
      }
    }
    #pragma unroll
    for (int nt = 0; nt < 4; ++nt) {
      int col = nt*16 + colB;
      float bb = b1s[col];
      #pragma unroll
      for (int rr = 0; rr < 8; ++rr) {
        float x = acc1[nt][rr] + bb;
        float h = x * fast_sigmoid(x);           // silu
        Hs[rbase + rr][col] = f2bf(h);
      }
    }
    __syncthreads();

    // ---- GEMM2: Y2(128x192) = Hs(128x64) @ W2(64x192), +b2 -> Y2s (aliases Xs/W1s) ----
    v8f acc2[12];
    #pragma unroll
    for (int nt = 0; nt < 12; ++nt)
      #pragma unroll
      for (int rr = 0; rr < 8; ++rr) acc2[nt][rr] = 0.f;

    for (int kc = 0; kc < 2; ++kc) {
      union { uint4 u[2]; v16bf v; } A;
      A.u[0] = *(const uint4*)&Hs[rowA][kc*32 + khalfA];
      A.u[1] = *(const uint4*)&Hs[rowA][kc*32 + khalfA + 16];
      #pragma unroll
      for (int nt = 0; nt < 12; ++nt) {
        union { uint4 u[2]; v16bf v; } B;
        B.u[0] = *(const uint4*)&W2s[nt*16 + colB][kc*32 + kbB];
        B.u[1] = *(const uint4*)&W2s[nt*16 + colB][kc*32 + kbB + 16];
        acc2[nt] = __builtin_amdgcn_wmma_f32_16x16x32_bf16(
            false, A.v, false, B.v, (short)0, acc2[nt], false, false);
      }
    }
    #pragma unroll
    for (int nt = 0; nt < 12; ++nt) {
      int col = nt*16 + colB;
      float bb = b2s[col];
      #pragma unroll
      for (int rr = 0; rr < 8; ++rr) Y2s[rbase + rr][col] = acc2[nt][rr] + bb;
    }
    __syncthreads();

    // ---- message assembly + atomic scatter (2 threads per edge) ----
    {
      int e = tid >> 1, half = tid & 1;
      EdgeInfo ei = Ed[e];
      const float* h  = &Y2s[e][0];                         // w00|wvs|wsv|wvvs|wvvv
      const float* kf = key_f + (size_t)ei.src * FDIM_;
      float* numrow   = num + (size_t)ei.dst * FDIM_;
      float dx = ei.dx, dy = ei.dy, dz = ei.dz;
      if (half == 0) atomicAdd(&cnt[ei.dst], 1.0f);
      for (int c = half; c < DV_; c += 2) {
        float ks_c  = kf[c];
        float ks_c2 = kf[DV_ + c];
        float kv0 = kf[DS_ + 3*c + 0], kv1 = kf[DS_ + 3*c + 1], kv2 = kf[DS_ + 3*c + 2];
        float dot = kv0*dx + kv1*dy + kv2*dz;
        float w00a = h[c], w00b = h[DV_ + c];
        float wvs  = h[64  + c];
        float wsv  = h[96  + c];
        float wvvs = h[128 + c];
        float wvvv = h[160 + c];
        float ms_a = w00a*ks_c + wvvs*dot;
        float ms_b = w00b*ks_c2;
        float cx = kv1*dz - kv2*dy;
        float cy = kv2*dx - kv0*dz;
        float cz = kv0*dy - kv1*dx;
        float mv0 = wvs*kv0 + wsv*ks_c*dx + wvvv*cx;
        float mv1 = wvs*kv1 + wsv*ks_c*dy + wvvv*cy;
        float mv2 = wvs*kv2 + wsv*ks_c*dz + wvvv*cz;
        atomicAdd(&numrow[c], ms_a);
        atomicAdd(&numrow[DV_ + c], ms_b);
        atomicAdd(&numrow[DS_ + 3*c + 0], mv0);
        atomicAdd(&numrow[DS_ + 3*c + 1], mv1);
        atomicAdd(&numrow[DS_ + 3*c + 2], mv2);
      }
    }
  }
}

// ---------------- tensor-product head: per n -> linq/angq ----------------
__global__ void k_head(const float* __restrict__ num0, const float* __restrict__ cnt0,
                       const float* __restrict__ num1, const float* __restrict__ cnt1,
                       const float* __restrict__ R, const float* __restrict__ query_f,
                       const float* __restrict__ p00, const float* __restrict__ prest,
                       const float* __restrict__ Ws_tp, const float* __restrict__ Wv_tp,
                       float* __restrict__ linq, float* __restrict__ angq)
{
  __shared__ float kf[FDIM_];
  __shared__ float qs[DS_];
  __shared__ float qvt[DV_][3];
  __shared__ float Rm[9];
  __shared__ float sarr[2][96];
  __shared__ float varr[2][96][3];
  __shared__ float accs[6];
  int n = blockIdx.x, t = n >> 8, q = n & 255, tid = threadIdx.x;

  if (tid < 9) Rm[tid] = R[t*9 + tid];
  if (tid < 6) accs[tid] = 0.f;
  __syncthreads();

  float ic0 = __builtin_amdgcn_rcpf(cnt0[n] + 1e-8f);
  float ic1 = __builtin_amdgcn_rcpf(cnt1[n] + 1e-8f);
  for (int j = tid; j < FDIM_; j += 64)
    kf[j] = num0[(size_t)n*FDIM_ + j]*ic0 + num1[(size_t)n*FDIM_ + j]*ic1;
  qs[tid] = query_f[(size_t)q*FDIM_ + tid];
  if (tid < DV_) {
    float v0 = query_f[(size_t)q*FDIM_ + DS_ + 3*tid + 0];
    float v1 = query_f[(size_t)q*FDIM_ + DS_ + 3*tid + 1];
    float v2 = query_f[(size_t)q*FDIM_ + DS_ + 3*tid + 2];
    qvt[tid][0] = Rm[0]*v0 + Rm[1]*v1 + Rm[2]*v2;
    qvt[tid][1] = Rm[3]*v0 + Rm[4]*v1 + Rm[5]*v2;
    qvt[tid][2] = Rm[6]*v0 + Rm[7]*v1 + Rm[8]*v2;
  }
  __syncthreads();

  for (int i = 0; i < 2; ++i) {
    for (int c = tid; c < 96; c += 64) {
      float s, v0, v1, v2;
      if (c < 64) {
        s = p00[i*64 + c] * kf[c] * qs[c];
      } else {
        int cc = c - 64;
        float k0 = kf[DS_+3*cc], k1 = kf[DS_+3*cc+1], k2 = kf[DS_+3*cc+2];
        s = prest[(i*4+2)*32 + cc] * (k0*qvt[cc][0] + k1*qvt[cc][1] + k2*qvt[cc][2]);
      }
      if (c < 32) {
        float k0 = kf[DS_+3*c], k1 = kf[DS_+3*c+1], k2 = kf[DS_+3*c+2];
        float p = prest[(i*4+0)*32 + c] * qs[c];
        v0 = p*k0; v1 = p*k1; v2 = p*k2;
      } else if (c < 64) {
        int cc = c - 32;
        float p = prest[(i*4+1)*32 + cc] * kf[cc];
        v0 = p*qvt[cc][0]; v1 = p*qvt[cc][1]; v2 = p*qvt[cc][2];
      } else {
        int cc = c - 64;
        float k0 = kf[DS_+3*cc], k1 = kf[DS_+3*cc+1], k2 = kf[DS_+3*cc+2];
        float p = prest[(i*4+3)*32 + cc];
        v0 = p*(k1*qvt[cc][2] - k2*qvt[cc][1]);
        v1 = p*(k2*qvt[cc][0] - k0*qvt[cc][2]);
        v2 = p*(k0*qvt[cc][1] - k1*qvt[cc][0]);
      }
      sarr[i][c] = s; varr[i][c][0] = v0; varr[i][c][1] = v1; varr[i][c][2] = v2;
    }
  }
  __syncthreads();

  int i = tid >> 5, o = tid & 31;
  float sg = 0.f;
  for (int c = 0; c < 96; ++c) sg += sarr[i][c] * Ws_tp[(i*96 + c)*33 + 1 + o];
  float g = fast_sigmoid(sg);
  float a0 = 0.f, a1 = 0.f, a2 = 0.f;
  for (int c = 0; c < 96; ++c) {
    float wv = Wv_tp[(i*96 + c)*32 + o];
    a0 += varr[i][c][0]*wv; a1 += varr[i][c][1]*wv; a2 += varr[i][c][2]*wv;
  }
  atomicAdd(&accs[i*3+0], a0*g);
  atomicAdd(&accs[i*3+1], a1*g);
  atomicAdd(&accs[i*3+2], a2*g);
  __syncthreads();
  if (tid < 6) {
    float val = accs[tid] * (1.f / (float)DV_);
    if (tid < 3) linq[(size_t)n*3 + tid] = val;
    else         angq[(size_t)n*3 + (tid-3)] = val;
  }
}

// ---------------- final per-frame reduction ----------------
__global__ void k_final(const float* __restrict__ R, const float* __restrict__ linq,
                        const float* __restrict__ angq, const float* __restrict__ query_x,
                        const float* __restrict__ query_w, float* __restrict__ out)
{
  __shared__ float Rm[9];
  __shared__ float red[6];
  int t = blockIdx.x, q = threadIdx.x, n = t*NQ_ + q;
  if (q < 9) Rm[q] = R[t*9 + q];
  if (q < 6) red[q] = 0.f;
  __syncthreads();
  float l0 = linq[(size_t)n*3+0], l1 = linq[(size_t)n*3+1], l2 = linq[(size_t)n*3+2];
  float a0 = angq[(size_t)n*3+0], a1 = angq[(size_t)n*3+1], a2 = angq[(size_t)n*3+2];
  // out_i = sum_j R[j][i] * v_j  (R^T v)
  float li0 = Rm[0]*l0 + Rm[3]*l1 + Rm[6]*l2;
  float li1 = Rm[1]*l0 + Rm[4]*l1 + Rm[7]*l2;
  float li2 = Rm[2]*l0 + Rm[5]*l1 + Rm[8]*l2;
  float ai0 = Rm[0]*a0 + Rm[3]*a1 + Rm[6]*a2;
  float ai1 = Rm[1]*a0 + Rm[4]*a1 + Rm[7]*a2;
  float ai2 = Rm[2]*a0 + Rm[5]*a1 + Rm[8]*a2;
  float x0 = query_x[q*3+0], x1 = query_x[q*3+1], x2 = query_x[q*3+2];
  float o0 = x1*li2 - x2*li1;
  float o1 = x2*li0 - x0*li2;
  float o2 = x0*li1 - x1*li0;
  float w = query_w[q];
  atomicAdd(&red[0], w*(o0 + ai0));
  atomicAdd(&red[1], w*(o1 + ai1));
  atomicAdd(&red[2], w*(o2 + ai2));
  atomicAdd(&red[3], w*li0);
  atomicAdd(&red[4], w*li1);
  atomicAdd(&red[5], w*li2);
  __syncthreads();
  if (q < 3)      out[t*3 + q] = red[q];              // ang_vel
  else if (q < 6) out[NT_*3 + t*3 + (q-3)] = red[q];  // lin_vel
}

// ---------------- launcher ----------------
extern "C" void kernel_launch(void* const* d_in, const int* in_sizes, int n_in,
                              void* d_out, int out_size, void* d_ws, size_t ws_size,
                              hipStream_t stream) {
  const float* Ts      = (const float*)d_in[0];
  const float* timev   = (const float*)d_in[1];
  const float* query_x = (const float*)d_in[2];
  const float* query_f = (const float*)d_in[3];
  const float* query_w = (const float*)d_in[4];
  const float* key_x0  = (const float*)d_in[5];
  const float* key_f0  = (const float*)d_in[6];
  const float* key_x1  = (const float*)d_in[7];
  const float* key_f1  = (const float*)d_in[8];
  const int*   es0     = (const int*)d_in[9];
  const int*   ed0     = (const int*)d_in[10];
  const int*   es1     = (const int*)d_in[11];
  const int*   ed1     = (const int*)d_in[12];
  const float* W_qt    = (const float*)d_in[13];
  const float* b_qt    = (const float*)d_in[14];
  const float* W1_r    = (const float*)d_in[15];
  const float* b1_r    = (const float*)d_in[16];
  const float* W2_r    = (const float*)d_in[17];
  const float* b2_r    = (const float*)d_in[18];
  const float* p00     = (const float*)d_in[19];
  const float* prest   = (const float*)d_in[20];
  const float* Ws_tp   = (const float*)d_in[21];
  const float* Wv_tp   = (const float*)d_in[22];

  char* ws = (char*)d_ws;
  float*          R      = (float*)(ws + OFF_R);
  float*          tr     = (float*)(ws + OFF_TR);
  unsigned short* qembbf = (unsigned short*)(ws + OFF_QEMB);
  float*          qxf    = (float*)(ws + OFF_QXF);
  float*          num0   = (float*)(ws + OFF_NUM0);
  float*          num1   = (float*)(ws + OFF_NUM1);
  float*          cnt0   = (float*)(ws + OFF_CNT0);
  float*          cnt1   = (float*)(ws + OFF_CNT1);
  unsigned short* w1t    = (unsigned short*)(ws + OFF_W1T);
  unsigned short* w2t    = (unsigned short*)(ws + OFF_W2T);
  float*          linq   = (float*)(ws + OFF_LINQ);
  float*          angq   = (float*)(ws + OFF_ANGQ);

  k_prep_R   <<<1, 64, 0, stream>>>(Ts, R, tr);
  k_prep_qemb<<<NT_, TE_, 0, stream>>>(timev, W_qt, b_qt, qembbf);
  k_prep_qx  <<<NN_/256, 256, 0, stream>>>(R, tr, query_x, qxf);
  k_prep_w1t <<<(2*HID_*KPAD_ + 255)/256, 256, 0, stream>>>(W1_r, w1t);
  k_prep_w2t <<<(2*HOUT_*HID_ + 255)/256, 256, 0, stream>>>(W2_r, w2t);
  k_zero     <<<2048, 256, 0, stream>>>(num0, ZERO_FLOATS);

  k_edge<<<EDGE_GRID_, 256, 0, stream>>>(es0, ed0, key_x0, key_f0, qxf, qembbf,
                                         w1t, b1_r, w2t, b2_r, num0, cnt0);
  k_edge<<<EDGE_GRID_, 256, 0, stream>>>(es1, ed1, key_x1, key_f1, qxf, qembbf,
                                         w1t + HID_*KPAD_, b1_r + HID_,
                                         w2t + HOUT_*HID_, b2_r + HOUT_, num1, cnt1);

  k_head <<<NN_, 64, 0, stream>>>(num0, cnt0, num1, cnt1, R, query_f,
                                  p00, prest, Ws_tp, Wv_tp, linq, angq);
  k_final<<<NT_, NQ_, 0, stream>>>(R, linq, angq, query_x, query_w, (float*)d_out);
}
